// PointnetFPModule_34651796144292
// MI455X (gfx1250) — matrix operand order repaired and checked by reading, not compiled, not run on hardware
//
#include <hip/hip_runtime.h>

typedef __attribute__((ext_vector_type(16))) _Float16 v16h;
typedef __attribute__((ext_vector_type(8)))  _Float16 v8h;
typedef __attribute__((ext_vector_type(2)))  _Float16 h2v;
typedef __attribute__((ext_vector_type(8)))  float    v8f;

// Problem sizes (fixed by reference)
// B=16, N=4096, M=1024, C1=128, C2=256, CIN=384, CMID=COUT=256

// ---------------- workspace layout (bytes) ----------------
// W1h   : 256*384 f16            @ 0        (196608)
// W2h   : 256*256 f16            @ 196608   (131072)
// sc1   : 256 f32                @ 327680
// sh1   : 256 f32                @ 328704
// sc2   : 256 f32                @ 329728
// sh2   : 256 f32                @ 330752
// kft   : 16*1024*256 f16        @ 331776   (8388608)   known_feats transposed
// idx3  : 16*4096*3 i32          @ 8720384  (786432)
// w3    : 16*4096*3 f32          @ 9506816  (786432)
// Xt    : 16*4096*384 f16        @ 10293248 (50331648)  concat activation, n-major
// total ~58 MB

__global__ void prep_params(const float* __restrict__ W1, const float* __restrict__ W2,
                            const float* __restrict__ g1, const float* __restrict__ b1,
                            const float* __restrict__ m1, const float* __restrict__ v1,
                            const float* __restrict__ g2, const float* __restrict__ b2,
                            const float* __restrict__ m2, const float* __restrict__ v2,
                            _Float16* __restrict__ W1h, _Float16* __restrict__ W2h,
                            float* __restrict__ sc1, float* __restrict__ sh1,
                            float* __restrict__ sc2, float* __restrict__ sh2)
{
    int i = blockIdx.x * 256 + threadIdx.x;
    if (i < 98304) {
        W1h[i] = (_Float16)W1[i];
    } else if (i < 163840) {
        int j = i - 98304;
        W2h[j] = (_Float16)W2[j];
    } else if (i < 164096) {
        int c = i - 163840;
        float s = g1[c] * rsqrtf(v1[c] + 1e-5f);
        sc1[c] = s; sh1[c] = b1[c] - m1[c] * s;
    } else if (i < 164352) {
        int c = i - 164096;
        float s = g2[c] * rsqrtf(v2[c] + 1e-5f);
        sc2[c] = s; sh2[c] = b2[c] - m2[c] * s;
    }
}

// (B, Rin, Cin) f32, channel-major -> out[b, col, outColOff + row] f16 (row stride = outStride)
__global__ void transpose_c_to_last(const float* __restrict__ in, _Float16* __restrict__ out,
                                    int Rin, int Cin, int outStride, int outColOff)
{
    __shared__ float tile[32][33];
    int bz = blockIdx.z;
    const float* inb = in + (size_t)bz * Rin * Cin;
    _Float16*   outb = out + (size_t)bz * Cin * outStride;
    int colBase = blockIdx.x * 32;
    int rowBase = blockIdx.y * 32;
#pragma unroll
    for (int i = 0; i < 4; ++i) {
        int r = rowBase + threadIdx.y + i * 8;
        tile[threadIdx.y + i * 8][threadIdx.x] = inb[(size_t)r * Cin + colBase + threadIdx.x];
    }
    __syncthreads();
#pragma unroll
    for (int i = 0; i < 4; ++i) {
        int oc = colBase + threadIdx.y + i * 8;  // original column -> out row
        outb[(size_t)oc * outStride + outColOff + rowBase + threadIdx.x] =
            (_Float16)tile[threadIdx.x][threadIdx.y + i * 8];
    }
}

__global__ void three_nn_kernel(const float* __restrict__ unknown, const float* __restrict__ known,
                                int* __restrict__ idx3, float* __restrict__ w3)
{
    __shared__ float4 sK[1024];
    int b = blockIdx.x >> 4;                        // 16 blocks per batch
    int n = ((blockIdx.x & 15) << 8) + threadIdx.x; // 256 points per block
    for (int j = threadIdx.x; j < 1024; j += 256) {
        const float* kp = known + ((size_t)b * 1024 + j) * 3;
        float kx = kp[0], ky = kp[1], kz = kp[2];
        sK[j] = make_float4(kx, ky, kz, kx * kx + ky * ky + kz * kz);
    }
    __syncthreads();
    const float* up = unknown + ((size_t)b * 4096 + n) * 3;
    float ux = up[0], uy = up[1], uz = up[2];
    float uu = ux * ux + uy * uy + uz * uz;
    float ax = -2.f * ux, ay = -2.f * uy, az = -2.f * uz;
    float d0 = 3e38f, d1 = 3e38f, d2 = 3e38f;
    int   i0 = 0, i1 = 0, i2 = 0;
    for (int m = 0; m < 1024; ++m) {
        float4 k = sK[m];  // uniform address -> broadcast
        float p = fmaf(ax, k.x, fmaf(ay, k.y, fmaf(az, k.z, k.w)));  // |u-k|^2 - |u|^2
        bool c0 = p < d0, c1 = p < d1, c2 = p < d2;
        float nd0 = c0 ? p : d0;
        float nd1 = c0 ? d0 : (c1 ? p : d1);
        float nd2 = c1 ? d1 : (c2 ? p : d2);
        int   ni0 = c0 ? m : i0;
        int   ni1 = c0 ? i0 : (c1 ? m : i1);
        int   ni2 = c1 ? i1 : (c2 ? m : i2);
        d0 = nd0; d1 = nd1; d2 = nd2; i0 = ni0; i1 = ni1; i2 = ni2;
    }
    float e0 = fmaxf(d0 + uu, 0.f), e1 = fmaxf(d1 + uu, 0.f), e2 = fmaxf(d2 + uu, 0.f);
    float r0 = 1.f / (e0 + 1e-8f), r1 = 1.f / (e1 + 1e-8f), r2 = 1.f / (e2 + 1e-8f);
    float inv = 1.f / (r0 + r1 + r2);
    size_t base = ((size_t)b * 4096 + n) * 3;
    idx3[base] = i0; idx3[base + 1] = i1; idx3[base + 2] = i2;
    w3[base] = r0 * inv; w3[base + 1] = r1 * inv; w3[base + 2] = r2 * inv;
}

__global__ void interp_kernel(const _Float16* __restrict__ kft, const int* __restrict__ idx3,
                              const float* __restrict__ w3, _Float16* __restrict__ Xt)
{
    int b = blockIdx.x >> 8;              // 256 blocks per batch (4096/16)
    int nbase = (blockIdx.x & 255) * 16;
    __shared__ int   sI[48];
    __shared__ float sW[48];
    if (threadIdx.x < 48) {
        size_t base = ((size_t)b * 4096 + nbase) * 3 + threadIdx.x;
        sI[threadIdx.x] = idx3[base];
        sW[threadIdx.x] = w3[base];
    }
    __syncthreads();
    int c = threadIdx.x;  // channel 0..255
    const _Float16* kb = kft + (size_t)b * 1024 * 256;
    for (int j = 0; j < 16; ++j) {
        int   i0 = sI[j * 3], i1 = sI[j * 3 + 1], i2 = sI[j * 3 + 2];
        float w0 = sW[j * 3], w1 = sW[j * 3 + 1], w2 = sW[j * 3 + 2];
        float v = w0 * (float)kb[(size_t)i0 * 256 + c]
                + w1 * (float)kb[(size_t)i1 * 256 + c]
                + w2 * (float)kb[(size_t)i2 * 256 + c];
        Xt[((size_t)b * 4096 + nbase + j) * 384 + c] = (_Float16)v;
    }
}

__device__ __forceinline__ v16h cat16(v8h lo, v8h hi)
{
    v16h r;
#pragma unroll
    for (int i = 0; i < 8; ++i) { r[i] = lo[i]; r[i + 8] = hi[i]; }
    return r;
}

// Fused: GEMM1(384->256) + BN + ReLU  ->  LDS  ->  GEMM2(256->256) + BN + ReLU -> d_out (fp32)
// One block per (batch, 32-wide n tile). 8 waves; each wave owns 32 output channels.
// K-outermost with 4 accumulators (2 o-tiles x 2 n-tiles) so each B fragment is
// loaded from LDS exactly once per K-step (4 WMMAs per 4 ds_load_b128).
__global__ __launch_bounds__(256) void fp_mlp_kernel(
    const _Float16* __restrict__ Xt,
    const _Float16* __restrict__ W1h, const float* __restrict__ sc1, const float* __restrict__ sh1,
    const _Float16* __restrict__ W2h, const float* __restrict__ sc2, const float* __restrict__ sh2,
    float* __restrict__ out)
{
    __shared__ __align__(32) _Float16 sX[32 * 384];  // staged input tile (n-major)
    __shared__ __align__(32) _Float16 sM[32 * 256];  // mid activation (n-major)

    int b  = blockIdx.x >> 7;        // 128 n-tiles per batch
    int n0 = (blockIdx.x & 127) * 32;
    int tid = threadIdx.x;

    // stage Xt tile: 32*384 halves = 1536 uint4
    {
        const uint4* src = reinterpret_cast<const uint4*>(Xt + ((size_t)b * 4096 + n0) * 384);
        uint4* dst = reinterpret_cast<uint4*>(sX);
        for (int i = tid; i < 1536; i += 256) dst[i] = src[i];
    }
    __syncthreads();

    int w = tid >> 5, lane = tid & 31;
    int m = lane & 15, g = lane >> 4;

    // ---------------- GEMM1: (256 x 384) x (384 x 32) ----------------
    {
        v8f acc00 = {}; v8f acc01 = {}; v8f acc10 = {}; v8f acc11 = {};
        for (int ks = 0; ks < 12; ++ks) {
            int c0 = ks * 32;
            // B fragments (shared by both o-tiles): lane(n=m,g) K = 16g..16g+15 contiguous
            v16h bb0 = *(const v16h*)(sX + (size_t)m * 384 + c0 + 16 * g);
            v16h bb1 = *(const v16h*)(sX + (size_t)(16 + m) * 384 + c0 + 16 * g);
            // A fragments: lane(m,g) K = 16g..16g+15 contiguous, o-tiles w*32 and w*32+16
            const _Float16* ar0 = W1h + (size_t)(w * 32 + m) * 384 + c0 + 16 * g;
            const _Float16* ar1 = ar0 + (size_t)16 * 384;
            v16h a0 = *(const v16h*)ar0;
            v16h a1 = *(const v16h*)ar1;
            acc00 = __builtin_amdgcn_wmma_f32_16x16x32_f16(false, a0, false, bb0, (short)0, acc00, false, false);
            acc01 = __builtin_amdgcn_wmma_f32_16x16x32_f16(false, a0, false, bb1, (short)0, acc01, false, false);
            acc10 = __builtin_amdgcn_wmma_f32_16x16x32_f16(false, a1, false, bb0, (short)0, acc10, false, false);
            acc11 = __builtin_amdgcn_wmma_f32_16x16x32_f16(false, a1, false, bb1, (short)0, acc11, false, false);
        }
        // D element (VGPR r, lane 16g+j) = (M=r+8g, N=j); BN+ReLU -> f16 pairs into sM
        auto epi1 = [&](v8f aN0, v8f aN1, int o0) {
            int oB = o0 + 8 * g;
#pragma unroll
            for (int r = 0; r < 8; r += 2) {
                float x0 = fmaxf(fmaf(aN0[r],     sc1[oB + r],     sh1[oB + r]),     0.f);
                float x1 = fmaxf(fmaf(aN0[r + 1], sc1[oB + r + 1], sh1[oB + r + 1]), 0.f);
                h2v p = { (_Float16)x0, (_Float16)x1 };
                *reinterpret_cast<h2v*>(&sM[(size_t)m * 256 + oB + r]) = p;
                float y0 = fmaxf(fmaf(aN1[r],     sc1[oB + r],     sh1[oB + r]),     0.f);
                float y1 = fmaxf(fmaf(aN1[r + 1], sc1[oB + r + 1], sh1[oB + r + 1]), 0.f);
                h2v q = { (_Float16)y0, (_Float16)y1 };
                *reinterpret_cast<h2v*>(&sM[(size_t)(16 + m) * 256 + oB + r]) = q;
            }
        };
        epi1(acc00, acc01, w * 32);
        epi1(acc10, acc11, w * 32 + 16);
    }
    __syncthreads();

    // ---------------- GEMM2: (256 x 256) x (256 x 32) ----------------
    {
        v8f acc00 = {}; v8f acc01 = {}; v8f acc10 = {}; v8f acc11 = {};
        for (int ks = 0; ks < 8; ++ks) {
            int c0 = ks * 32;
            v16h bb0 = *(const v16h*)(sM + (size_t)m * 256 + c0 + 16 * g);
            v16h bb1 = *(const v16h*)(sM + (size_t)(16 + m) * 256 + c0 + 16 * g);
            const _Float16* ar0 = W2h + (size_t)(w * 32 + m) * 256 + c0 + 16 * g;
            const _Float16* ar1 = ar0 + (size_t)16 * 256;
            v16h a0 = *(const v16h*)ar0;
            v16h a1 = *(const v16h*)ar1;
            acc00 = __builtin_amdgcn_wmma_f32_16x16x32_f16(false, a0, false, bb0, (short)0, acc00, false, false);
            acc01 = __builtin_amdgcn_wmma_f32_16x16x32_f16(false, a0, false, bb1, (short)0, acc01, false, false);
            acc10 = __builtin_amdgcn_wmma_f32_16x16x32_f16(false, a1, false, bb0, (short)0, acc10, false, false);
            acc11 = __builtin_amdgcn_wmma_f32_16x16x32_f16(false, a1, false, bb1, (short)0, acc11, false, false);
        }
        auto epi2 = [&](v8f aN0, v8f aN1, int o0) {
            int oB = o0 + 8 * g;
            float* ob = out + ((size_t)b * 256 + oB) * 4096 + n0;
#pragma unroll
            for (int r = 0; r < 8; ++r) {
                ob[(size_t)r * 4096 + m]      = fmaxf(fmaf(aN0[r], sc2[oB + r], sh2[oB + r]), 0.f);
                ob[(size_t)r * 4096 + 16 + m] = fmaxf(fmaf(aN1[r], sc2[oB + r], sh2[oB + r]), 0.f);
            }
        };
        epi2(acc00, acc01, w * 32);
        epi2(acc10, acc11, w * 32 + 16);
    }
}

extern "C" void kernel_launch(void* const* d_in, const int* in_sizes, int n_in,
                              void* d_out, int out_size, void* d_ws, size_t ws_size,
                              hipStream_t stream)
{
    const float* unknown = (const float*)d_in[0];
    const float* known   = (const float*)d_in[1];
    const float* uf      = (const float*)d_in[2];   // (16,128,4096)
    const float* kf      = (const float*)d_in[3];   // (16,256,1024)
    const float* W1 = (const float*)d_in[4];
    const float* g1 = (const float*)d_in[5];
    const float* b1 = (const float*)d_in[6];
    const float* m1 = (const float*)d_in[7];
    const float* v1 = (const float*)d_in[8];
    const float* W2 = (const float*)d_in[9];
    const float* g2 = (const float*)d_in[10];
    const float* b2 = (const float*)d_in[11];
    const float* m2 = (const float*)d_in[12];
    const float* v2 = (const float*)d_in[13];

    char* ws = (char*)d_ws;
    _Float16* W1h = (_Float16*)(ws + 0);
    _Float16* W2h = (_Float16*)(ws + 196608);
    float* sc1 = (float*)(ws + 327680);
    float* sh1 = (float*)(ws + 328704);
    float* sc2 = (float*)(ws + 329728);
    float* sh2 = (float*)(ws + 330752);
    _Float16* kft = (_Float16*)(ws + 331776);
    int*   idx3 = (int*)(ws + 8720384);
    float* w3   = (float*)(ws + 9506816);
    _Float16* Xt = (_Float16*)(ws + 10293248);
    float* out = (float*)d_out;

    prep_params<<<642, 256, 0, stream>>>(W1, W2, g1, b1, m1, v1, g2, b2, m2, v2,
                                         W1h, W2h, sc1, sh1, sc2, sh2);
    // known_feats (16,256,1024) -> kft (16,1024,256) f16
    transpose_c_to_last<<<dim3(32, 8, 16), dim3(32, 8), 0, stream>>>(kf, kft, 256, 1024, 256, 0);
    // unknow_feats (16,128,4096) -> Xt[:, :, 256:384] f16
    transpose_c_to_last<<<dim3(128, 4, 16), dim3(32, 8), 0, stream>>>(uf, Xt, 128, 4096, 384, 256);
    three_nn_kernel<<<256, 256, 0, stream>>>(unknown, known, idx3, w3);
    interp_kernel<<<4096, 256, 0, stream>>>(kft, idx3, w3, Xt);
    fp_mlp_kernel<<<2048, 256, 0, stream>>>(Xt, W1h, sc1, sh1, W2h, sc2, sh2, out);
}
